// VIN_19327352832072
// MI455X (gfx1250) — matrix operand order repaired and checked by reading, not compiled, not run on hardware
//
#include <hip/hip_runtime.h>
#include <hip/hip_bf16.h>

typedef __attribute__((ext_vector_type(16))) _Float16 v16h;
typedef __attribute__((ext_vector_type(8)))  float    v8f;

#define BATCH 128
#define IMS   64
#define PADW  66            // 64 + halo
#define ROWS  198           // [r | v0 | v1] slots per padded row (f16)
#define LQ    10
#define LI    2
#define LH    150
#define NACT  8
#define QSIZE (BATCH * LQ * IMS * IMS)
#define NEGINF (-3.402823466e38f)

typedef __attribute__((address_space(1))) int glb_i32;
typedef __attribute__((address_space(3))) int lds_i32;

// ---- CDNA5 async global->LDS copy (builtin present; param0 = global int*) ----
#if __has_builtin(__builtin_amdgcn_global_load_async_to_lds_b32)
#define HAVE_ASYNC_LDS 1
__device__ __forceinline__ void async_cp32(const float* g, float* l) {
  __builtin_amdgcn_global_load_async_to_lds_b32(
      (glb_i32*)g, (lds_i32*)l, 0, 0);
}
#else
#define HAVE_ASYNC_LDS 0
__device__ __forceinline__ void async_cp32(const float* g, float* l) { *l = *g; }
#endif

__device__ __forceinline__ void wait_async0() {
#if HAVE_ASYNC_LDS
#if __has_builtin(__builtin_amdgcn_s_wait_asynccnt)
  __builtin_amdgcn_s_wait_asynccnt(0);
#else
  asm volatile("s_wait_asynccnt 0" ::: "memory");
#endif
#endif
}

// One 16-pixel tile eval: D(16x16) = A(16x32 f16 weights) x B(32x16 f16 patch).
// K mapping: k = dy*6 + c*3 + dx  (c: 0=r, 1=v), k<18 real, rest zero-padded.
// Assumed B 32x16 f16 layout: lanes 0-15 hold K {0..7,16..23}; lanes 16-31 -> +8.
__device__ __forceinline__ v8f eval_tile(const _Float16* __restrict__ rvh,
                                         int y, int X, int voff,
                                         v16h afrag, int half) {
  const _Float16* baseR = rvh + y * ROWS + X;    // r slot (f16, const ds offsets)
  const _Float16* baseV = baseR + voff;          // active v slot
  _Float16 vals[18];
  #pragma unroll
  for (int dy = 0; dy < 3; ++dy) {
    #pragma unroll
    for (int dx = 0; dx < 3; ++dx) {
      vals[dy * 6 + dx]     = baseR[dy * ROWS + dx];
      vals[dy * 6 + 3 + dx] = baseV[dy * ROWS + dx];
    }
  }
  v16h bfrag;
  #pragma unroll
  for (int e = 0; e < 16; ++e) {
    _Float16 lo = (e < 8) ? vals[e] : ((e < 10) ? vals[e + 8] : (_Float16)0.f);   // half==0: K=e / e+8
    _Float16 hi = (e < 8) ? vals[8 + e] : (_Float16)0.f;    // half==1: K=8+e / e+16
    bfrag[e] = half ? hi : lo;
  }
  v8f acc = {};
  return __builtin_amdgcn_wmma_f32_16x16x32_f16(
      /*neg_a=*/false, afrag, /*neg_b=*/false, bfrag,
      /*c_mod=*/(short)0, acc, /*reuse_a=*/false, /*reuse_b=*/false);
}

// One block = one batch image; 256 threads = 8 wave32s.
__global__ __launch_bounds__(256) void vin_iter_kernel(
    const float* __restrict__ input_view,   // (B,2,64,64)
    const int*   __restrict__ state_x,
    const int*   __restrict__ state_y,
    const int*   __restrict__ kptr,
    const float* __restrict__ h_w,          // (150,2,3,3)
    const float* __restrict__ h_b,          // (150,)
    const float* __restrict__ r_w,          // (150,)
    const float* __restrict__ q_w,          // (10,1,3,3)
    const float* __restrict__ w,            // (10,1,3,3)
    const float* __restrict__ fc_w,         // (8,10)
    float* __restrict__ out)                // q | logits | action
{
  __shared__ _Float16 rvh[PADW * ROWS];         // f16 [r|v0|v1] rows, ~26 KB
  __shared__ float    stage[LI * PADW * PADW];  // f32 staged input, ~35 KB
  __shared__ float    effw[18];
  __shared__ float    effb;
  __shared__ float    qcol[16];

  const int b    = blockIdx.x;
  const int tid  = threadIdx.x;
  const int lane = tid & 31;
  const int wave = tid >> 5;
  const int n    = lane & 15;
  const int half = lane >> 4;

  const float* inp = input_view + (size_t)b * LI * IMS * IMS;

  // Zero LDS (padded borders stay zero forever).
  for (int i = tid; i < PADW * ROWS; i += 256) rvh[i] = (_Float16)0.f;
  for (int i = tid; i < LI * PADW * PADW; i += 256) stage[i] = 0.f;
  // Fused h->r path: eff 2ch 3x3 conv, effw[c*9+off] = sum_ch r_w*h_w.
  if (tid < 18) {
    int ci = tid / 9, off = tid % 9;
    float s = 0.f;
    for (int c = 0; c < LH; ++c) s += r_w[c] * h_w[(c * LI + ci) * 9 + off];
    effw[tid] = s;
  }
  if (tid == 18) {
    float s = 0.f;
    for (int c = 0; c < LH; ++c) s += r_w[c] * h_b[c];
    effb = s;
  }
  if (tid < 16) qcol[tid] = 0.f;
  __syncthreads();

  // Async-stage both input channels into padded f32 scratch.
  for (int idx = tid; idx < LI * IMS * IMS; idx += 256) {
    int ci  = idx >> 12;
    int rem = idx & 4095;
    int row = rem >> 6, col = rem & 63;
    async_cp32(inp + idx, &stage[ci * PADW * PADW + (row + 1) * PADW + (col + 1)]);
  }
  wait_async0();
  __syncthreads();

  // r = conv3x3(input, effw) + effb, branchless, stored as f16 into r slot.
  for (int p = tid; p < IMS * IMS; p += 256) {
    int y = p >> 6, x = p & 63;
    const float* b1 = stage + y * PADW + x;            // channel 0 window
    const float* b2 = b1 + PADW * PADW;                // channel 1 window
    float acc = effb;
    #pragma unroll
    for (int dy = 0; dy < 3; ++dy)
      #pragma unroll
      for (int dx = 0; dx < 3; ++dx) {
        acc += effw[dy * 3 + dx]     * b1[dy * PADW + dx];
        acc += effw[9 + dy * 3 + dx] * b2[dy * PADW + dx];
      }
    rvh[(y + 1) * ROWS + (x + 1)] = (_Float16)acc;
  }

  // Constant A fragment with k = dy*6 + c*3 + dx mapping.
  // 16-bit A 16x32 layout: lanes 0-15 -> K {0..7,16..23}; lanes 16-31 -> +8.
  v16h afrag;
  #pragma unroll
  for (int e = 0; e < 16; ++e) {
    int j = e >> 1, pos = e & 1;
    int k = ((j < 4) ? (2 * j + pos) : (16 + 2 * (j - 4) + pos)) + half * 8;
    float wv = 0.f;
    if (n < LQ && k < 18) {
      int dy = k / 6, rem = k % 6;
      int ci = rem / 3, dx = rem % 3;
      wv = (ci == 0 ? q_w : w)[n * 9 + dy * 3 + dx];
    }
    afrag[e] = (_Float16)wv;
  }
  __syncthreads();

  const int K  = kptr[0];
  const int sx = state_x[b];
  const int sy = state_y[b];

  // K sweeps of v = max_m conv(concat(r,v)); t=0 reads v==0 (the q0 path).
  for (int t = 0; t < K; ++t) {
    const int vin  = 66 * (1 + (t & 1));
    const int vout = 66 * (1 + ((t + 1) & 1));
    for (int tt = 0; tt < 32; ++tt) {
      int tile = wave * 32 + tt;
      int y = tile >> 2, X = ((tile & 3) << 4) + n;
      v8f acc = eval_tile(rvh, y, X, vin, afrag, half);
      // max over m: lanes 0-15 hold m=j, lanes 16-31 m=j+8 (only m<10 real).
      float mx = fmaxf(acc[0], acc[1]);
      #pragma unroll
      for (int j = 2; j < 8; ++j) mx = fmaxf(mx, half ? NEGINF : acc[j]);
      mx = fmaxf(mx, __shfl_xor(mx, 16, 32));
      // Both halves hold the identical combined max: same-value store, no branch.
      rvh[(y + 1) * ROWS + vout + (X + 1)] = (_Float16)mx;
    }
    __syncthreads();   // ping-pong => one barrier per sweep
  }

  // Final eval: q = conv(concat(r, v)); store q, capture (sx,sy) column.
  {
    const int vin = 66 * (1 + (K & 1));
    for (int tt = 0; tt < 32; ++tt) {
      int tile = wave * 32 + tt;
      int y = tile >> 2, X = ((tile & 3) << 4) + n;
      v8f acc = eval_tile(rvh, y, X, vin, afrag, half);
      #pragma unroll
      for (int j = 0; j < 8; ++j) {
        int m = j + half * 8;
        if (m < LQ)
          out[(((size_t)b * LQ + m) * IMS + y) * IMS + X] = acc[j];
      }
      if (y == sx && X == sy) {
        #pragma unroll
        for (int j = 0; j < 8; ++j) qcol[j + half * 8] = acc[j];
      }
    }
    __syncthreads();
  }

  // logits[b,:] = qcol @ fc_w.T
  if (tid < NACT) {
    float s = 0.f;
    #pragma unroll
    for (int a = 0; a < LQ; ++a) s += qcol[a] * fc_w[tid * LQ + a];
    out[QSIZE + b * NACT + tid] = s;
  }
}

// Flat argmax over 128x8 logits -> action (stored as float).
__global__ __launch_bounds__(1024) void vin_argmax_kernel(
    const float* __restrict__ logits, float* __restrict__ action)
{
  __shared__ float vals[BATCH * NACT];
  int tid = threadIdx.x;
  vals[tid] = logits[tid];
  __syncthreads();
  if (tid == 0) {
    float best = vals[0];
    int bi = 0;
    for (int i = 1; i < BATCH * NACT; ++i)
      if (vals[i] > best) { best = vals[i]; bi = i; }
    *action = (float)bi;
  }
}

extern "C" void kernel_launch(void* const* d_in, const int* in_sizes, int n_in,
                              void* d_out, int out_size, void* d_ws, size_t ws_size,
                              hipStream_t stream) {
  (void)in_sizes; (void)n_in; (void)d_ws; (void)ws_size; (void)out_size;
  const float* input_view = (const float*)d_in[0];
  const int*   sx         = (const int*)  d_in[1];
  const int*   sy         = (const int*)  d_in[2];
  const int*   kptr       = (const int*)  d_in[3];
  const float* h_w        = (const float*)d_in[4];
  const float* h_b        = (const float*)d_in[5];
  const float* r_w        = (const float*)d_in[6];
  const float* q_w        = (const float*)d_in[7];
  const float* w          = (const float*)d_in[8];
  const float* fc_w       = (const float*)d_in[9];
  float* out = (float*)d_out;

  vin_iter_kernel<<<BATCH, 256, 0, stream>>>(input_view, sx, sy, kptr,
                                             h_w, h_b, r_w, q_w, w, fc_w, out);
  vin_argmax_kernel<<<1, BATCH * NACT, 0, stream>>>(out + QSIZE,
                                                    out + QSIZE + BATCH * NACT);
}
